// Decoder_38397007626387
// MI455X (gfx1250) — matrix operand order repaired and checked by reading, not compiled, not run on hardware
//
#include <hip/hip_runtime.h>
#include <hip/hip_bf16.h>
#include <stdint.h>

// ---------------------------------------------------------------------------
// Heterogeneous GAT decoder for MI455X (gfx1250, wave32, WMMA).
// Dominant GEMMs run as bf16 WMMA (v_wmma_f32_16x16x32_bf16) with f32 accum.
// Fast GEMM (K%32==0, N%8==0): 256x64 block tile, 8 WMMA/wave/k-step,
// software-pipelined unconditional global_load_b128 staging into *named*
// scalar registers (no alloca -> no LDS bounce), fragment-ordered LDS so each
// WMMA operand is a contiguous 32B ds_load_b128 pair.
// zd / ssrc are algebraically folded (W @ a vectors) so only zs is stored.
// ---------------------------------------------------------------------------

#define NFN 40000
#define NRN 4000
#define RDIM 14
#define NE_FF 300000
#define NE_RR 40000
#define NE_RF 120000

typedef __attribute__((ext_vector_type(16))) __bf16 v16bf;
typedef __attribute__((ext_vector_type(8)))  float  v8f;

static __host__ __device__ inline int ceilDiv(long a, long b) { return (int)((a + b - 1) / b); }

// ---------------- elementwise kernels ----------------
__global__ void reparam_kernel(const float* __restrict__ mu, const float* __restrict__ lv,
                               const float* __restrict__ ep, float* __restrict__ out, size_t n) {
  size_t i = (size_t)blockIdx.x * blockDim.x + threadIdx.x;
  if (i < n) out[i] = mu[i] + __expf(0.5f * lv[i]) * ep[i];
}

__global__ void f32_to_bf16_kernel(const float* __restrict__ x, __bf16* __restrict__ y, size_t n) {
  size_t i = (size_t)blockIdx.x * blockDim.x + threadIdx.x;
  if (i < n) y[i] = (__bf16)x[i];
}

__global__ void fill_f32_kernel(float* __restrict__ p, float v, size_t n) {
  size_t i = (size_t)blockIdx.x * blockDim.x + threadIdx.x;
  if (i < n) p[i] = v;
}

__global__ void fill_u32_kernel(unsigned* __restrict__ p, unsigned v, size_t n) {
  size_t i = (size_t)blockIdx.x * blockDim.x + threadIdx.x;
  if (i < n) p[i] = v;
}

__global__ void elu_ip_kernel(float* __restrict__ x, size_t n) {
  size_t i = (size_t)blockIdx.x * blockDim.x + threadIdx.x;
  if (i < n) { float v = x[i]; x[i] = v > 0.f ? v : (__expf(v) - 1.f); }
}

// ---------------- attention-vector fold: wv[i] = dot(W[i,:], a); wv[fin] = dot(b, a) ----------------
__global__ void fold_attn_kernel(const float* __restrict__ W, const float* __restrict__ a,
                                 const float* __restrict__ b, int fin, int fout,
                                 float* __restrict__ wv) {
  int i = blockIdx.x * blockDim.x + threadIdx.x;
  if (i > fin) return;
  const float* row = (i < fin) ? (W + (size_t)i * fout) : b;
  float v = 0.f;
  for (int j = 0; j < fout; ++j) v += row[j] * a[j];
  wv[i] = v;
}

// ---------------- per-node score: s[n] = dot(h[n], wv[:fin]) + wv[fin]; one wave32 per node ----------------
__global__ __launch_bounds__(256) void node_score_kernel(const float* __restrict__ H,
                                                         const float* __restrict__ wv,
                                                         int fin, int Nn, float* __restrict__ out) {
  int wave = threadIdx.x >> 5, lane = threadIdx.x & 31;
  int n = blockIdx.x * 8 + wave;
  if (n >= Nn) return;
  const float* row = H + (size_t)n * fin;
  float v = 0.f;
  for (int k = lane; k < fin; k += 32) v += row[k] * wv[k];
  for (int off = 16; off; off >>= 1) v += __shfl_xor(v, off, 32);
  if (lane == 0) out[n] = v + wv[fin];
}

// ---------------- WMMA GEMM fragment-order helpers ----------------
// A elem (mm,kk): block = (mm>>4)*32 + ((kk>>3)&1)*16 + (mm&15); pos = (kk&7)|((kk&16)>>1)
// B elem (kk,nn): block = (nn>>4)*32 + ((kk>>3)&1)*16 + (nn&15); pos = (kk&7)|((kk&16)>>1)
#define GKS 32
#define GNT 64

// ================== FAST GEMM: requires K%32==0 && N%8==0 ==================
#define FMT 256

__global__ __launch_bounds__(256) void gemm_bf16_wmma_fast_kernel(
    const __bf16* __restrict__ A, const __bf16* __restrict__ B,
    const float* __restrict__ bias, float* __restrict__ C,
    int M, int N, int K) {
  __shared__ __bf16 sAf[FMT * GKS];       // 8192 bf16 = 16KB, fragment-ordered
  __shared__ __bf16 sBf[4 * 32 * 16];     // 2048 bf16 = 4KB, fragment-ordered
  const int tid = threadIdx.x;
  const int wave = tid >> 5;
  const int lane = tid & 31;
  const int lrow = lane & 15;
  const int hi   = lane >> 4;
  const int m0 = blockIdx.y * FMT;
  const int n0 = blockIdx.x * GNT;

  // per-thread A staging: 4 chunks of 8 bf16 (clamped source rows, fixed LDS dests)
#define A_SETUP(CC, SRCV, DSTV)                                              \
  const __bf16* SRCV; __bf16* DSTV;                                          \
  {                                                                          \
    int c = tid + (CC) * 256;                                                \
    int mm = c >> 2, kch = (c & 3) * 8;                                      \
    int gm = m0 + mm;                                                        \
    int gmc = gm < M ? gm : (M - 1);   /* OOB rows masked in epilogue */     \
    SRCV = A + (size_t)gmc * K + kch;                                        \
    int hi2 = (kch >> 3) & 1, ib = (kch & 16) >> 1;                          \
    DSTV = &sAf[(((mm >> 4) * 32 + hi2 * 16 + (mm & 15)) << 4) + ib];        \
  }
  A_SETUP(0, aS0, aD0)
  A_SETUP(1, aS1, aD1)
  A_SETUP(2, aS2, aD2)
  A_SETUP(3, aS3, aD3)
#undef A_SETUP

  // per-thread B staging: 1 chunk of 8 bf16, scattered to 8 fragment slots
  const int bkk = tid >> 3;
  const int bnnb = (tid & 7) * 8;
  int gn0 = n0 + bnnb;
  int gnc = gn0 < N ? gn0 : (N - 8);          // OOB cols masked in epilogue
  const __bf16* bS = B + (size_t)bkk * N + gnc;
  const size_t bStep = (size_t)GKS * N;
  const int bhi2 = (bkk >> 3) & 1, bke = (bkk & 7) | ((bkk & 16) >> 1);
  __bf16* bD0; __bf16* bD1; __bf16* bD2; __bf16* bD3;
  __bf16* bD4; __bf16* bD5; __bf16* bD6; __bf16* bD7;
  {
#define B_DST(Q) &sBf[(((((bnnb + (Q)) >> 4) * 32) + bhi2 * 16 + ((bnnb + (Q)) & 15)) << 4) + bke]
    bD0 = B_DST(0); bD1 = B_DST(1); bD2 = B_DST(2); bD3 = B_DST(3);
    bD4 = B_DST(4); bD5 = B_DST(5); bD6 = B_DST(6); bD7 = B_DST(7);
#undef B_DST
  }

  // staging registers: named scalars so they never become an alloca
  uint4 ra0, ra1, ra2, ra3, rb;

  v8f acc0[4] = {};
  v8f acc1[4] = {};

  ra0 = *(const uint4*)aS0;
  ra1 = *(const uint4*)aS1;
  ra2 = *(const uint4*)aS2;
  ra3 = *(const uint4*)aS3;
  rb  = *(const uint4*)bS;

  for (int k0 = 0; k0 < K; k0 += GKS) {
    // deposit staged registers into fragment-ordered LDS
    *(uint4*)aD0 = ra0;
    *(uint4*)aD1 = ra1;
    *(uint4*)aD2 = ra2;
    *(uint4*)aD3 = ra3;
    {
      union { uint4 v; unsigned u[4]; } bu; bu.v = rb;
      __bf16 b2[2];
      *(unsigned*)b2 = bu.u[0]; *bD0 = b2[0]; *bD1 = b2[1];
      *(unsigned*)b2 = bu.u[1]; *bD2 = b2[0]; *bD3 = b2[1];
      *(unsigned*)b2 = bu.u[2]; *bD4 = b2[0]; *bD5 = b2[1];
      *(unsigned*)b2 = bu.u[3]; *bD6 = b2[0]; *bD7 = b2[1];
    }
    __syncthreads();
    if (k0 + GKS < K) {   // issue next-tile HBM loads; consumed only next iter
      aS0 += GKS; aS1 += GKS; aS2 += GKS; aS3 += GKS; bS += bStep;
      ra0 = *(const uint4*)aS0;
      ra1 = *(const uint4*)aS1;
      ra2 = *(const uint4*)aS2;
      ra3 = *(const uint4*)aS3;
      rb  = *(const uint4*)bS;
    }
    v16bf a0 = *(const v16bf*)&sAf[((2 * wave + 0) * 32 + lane) << 4];
    v16bf a1 = *(const v16bf*)&sAf[((2 * wave + 1) * 32 + lane) << 4];
    #pragma unroll
    for (int t = 0; t < 4; ++t) {
      v16bf bf_ = *(const v16bf*)&sBf[(t * 32 + lane) << 4];
      acc0[t] = __builtin_amdgcn_wmma_f32_16x16x32_bf16(
          false, a0, false, bf_, (short)0, acc0[t], false, false);
      acc1[t] = __builtin_amdgcn_wmma_f32_16x16x32_bf16(
          false, a1, false, bf_, (short)0, acc1[t], false, false);
    }
    __syncthreads();
  }

  // epilogue: D layout — VGPR i: row M=i (lanes<16) / i+8 (lanes>=16), col N=lane&15
  #pragma unroll
  for (int t = 0; t < 4; ++t) {
    int gn = n0 + t * 16 + lrow;
    if (gn >= N) continue;
    float bb = bias ? bias[gn] : 0.f;
    #pragma unroll
    for (int i = 0; i < 8; ++i) {
      int gm0 = m0 + wave * 32 + i + hi * 8;
      if (gm0 < M)      C[(size_t)gm0 * N + gn] = acc0[t][i] + bb;
      int gm1 = gm0 + 16;
      if (gm1 < M)      C[(size_t)gm1 * N + gn] = acc1[t][i] + bb;
    }
  }
}

// ================== GENERIC GEMM: any K, N (small room GEMMs) ==================
#define GMT 128

__global__ __launch_bounds__(256) void gemm_bf16_wmma_kernel(
    const __bf16* __restrict__ A, const __bf16* __restrict__ B,
    const float* __restrict__ bias, float* __restrict__ C,
    int M, int N, int K) {
  __shared__ __bf16 sAf[GMT * GKS];
  __shared__ __bf16 sBf[4 * 32 * 16];
  const int tid = threadIdx.x;
  const int wave = tid >> 5;
  const int lane = tid & 31;
  const int lrow = lane & 15;
  const int hi   = lane >> 4;
  const int m0 = blockIdx.y * GMT;
  const int n0 = blockIdx.x * GNT;

  v8f acc[4] = {};

  for (int k0 = 0; k0 < K; k0 += GKS) {
    #pragma unroll
    for (int cc = 0; cc < 2; ++cc) {
      int c = tid + cc * 256;
      int mm = c >> 2;
      int kch = (c & 3) * 8;
      int gm = m0 + mm, gk = k0 + kch;
      int hi2 = (kch >> 3) & 1;
      int ib = (kch & 16) >> 1;
      __bf16* dst8 = &sAf[(((mm >> 4) * 32 + hi2 * 16 + (mm & 15)) << 4) + ib];
      #pragma unroll
      for (int q = 0; q < 8; ++q) {
        int gkq = gk + q;
        dst8[q] = (gm < M && gkq < K) ? A[(size_t)gm * K + gkq] : (__bf16)0.0f;
      }
    }
    {
      int kk = tid >> 3;
      int nnb = (tid & 7) * 8;
      int gk = k0 + kk, gn0 = n0 + nnb;
      __bf16 vals[8] __attribute__((aligned(16)));
      #pragma unroll
      for (int q = 0; q < 8; ++q) {
        int gn = gn0 + q;
        vals[q] = (gk < K && gn < N) ? B[(size_t)gk * N + gn] : (__bf16)0.0f;
      }
      int hi2 = (kk >> 3) & 1;
      int ke = (kk & 7) | ((kk & 16) >> 1);
      #pragma unroll
      for (int q = 0; q < 8; ++q) {
        int nn = nnb + q;
        sBf[((((nn >> 4) * 32) + hi2 * 16 + (nn & 15)) << 4) + ke] = vals[q];
      }
    }
    __syncthreads();

    v16bf afrag = *(const v16bf*)&sAf[(wave * 32 + lane) << 4];
    #pragma unroll
    for (int t = 0; t < 4; ++t) {
      v16bf bfrag = *(const v16bf*)&sBf[(t * 32 + lane) << 4];
      acc[t] = __builtin_amdgcn_wmma_f32_16x16x32_bf16(
          false, afrag, false, bfrag, (short)0, acc[t], false, false);
    }
    __syncthreads();
  }

  #pragma unroll
  for (int t = 0; t < 4; ++t) {
    #pragma unroll
    for (int i = 0; i < 8; ++i) {
      int gm = m0 + wave * 16 + i + (hi ? 8 : 0);
      int gn = n0 + t * 16 + lrow;
      if (gm < M && gn < N)
        C[(size_t)gm * N + gn] = acc[t][i] + (bias ? bias[gn] : 0.f);
    }
  }
}

// ---------------- segment softmax passes ----------------
__device__ inline unsigned enc_f32(float x) {
  unsigned u = __float_as_uint(x);
  return (u & 0x80000000u) ? ~u : (u | 0x80000000u);
}
__device__ inline float dec_f32(unsigned u) {
  unsigned v = (u & 0x80000000u) ? (u & 0x7FFFFFFFu) : ~u;
  return __uint_as_float(v);
}

__global__ void edge_pass1_kernel(const float* __restrict__ ssrc, const float* __restrict__ sdst,
                                  const int* __restrict__ src, const int* __restrict__ dst,
                                  int E, float* __restrict__ logit, unsigned* __restrict__ maxb) {
  int e = blockIdx.x * blockDim.x + threadIdx.x;
  if (e >= E) return;
  float l = ssrc[src[e]] + sdst[dst[e]];
  l = l > 0.f ? l : 0.2f * l;  // leaky_relu 0.2
  logit[e] = l;
  atomicMax(&maxb[dst[e]], enc_f32(l));
}

__global__ void edge_pass2_kernel(const float* __restrict__ logit, const unsigned* __restrict__ maxb,
                                  const int* __restrict__ dst, int E,
                                  float* __restrict__ wexp, float* __restrict__ sumb) {
  int e = blockIdx.x * blockDim.x + threadIdx.x;
  if (e >= E) return;
  float m = dec_f32(maxb[dst[e]]);
  if (!(m > -3.0e38f)) m = 0.f;  // match reference: non-finite max -> 0
  float w = __expf(logit[e] - m);
  wexp[e] = w;
  atomicAdd(&sumb[dst[e]], w);
}

// acc[dst] += alpha * zs[src]; one block per edge, threads stride features
__global__ void edge_scatter_kernel(const float* __restrict__ zs, const int* __restrict__ src,
                                    const int* __restrict__ dst, const float* __restrict__ wexp,
                                    const float* __restrict__ sumb, int E, int dout,
                                    float* __restrict__ acc) {
  int e = blockIdx.x;
  if (e >= E) return;
  int s = src[e], d = dst[e];
  float alpha = wexp[e] / (sumb[d] + 1e-9f);
  const float* zrow = zs + (size_t)s * dout;
  float* arow = acc + (size_t)d * dout;
  for (int f = threadIdx.x; f < dout; f += blockDim.x)
    atomicAdd(&arow[f], alpha * zrow[f]);
}

// ---------------- fused edge MLP: out = leaky_relu(concat(segments) @ We + be) ----------------
__global__ __launch_bounds__(256) void edge_mlp_kernel(
    const int* __restrict__ src, const int* __restrict__ dst, int E,
    const float* p0, int w0, int md0,
    const float* p1, int w1, int md1,
    const float* p2, int w2, int md2,
    const float* p3, int w3, int md3,
    const float* __restrict__ We, const float* __restrict__ be,
    int dein, int dout, float* __restrict__ out) {
  extern __shared__ float sW[];
  for (int i = threadIdx.x; i < dein * dout; i += blockDim.x) sW[i] = We[i];
  __syncthreads();
  int wave = threadIdx.x >> 5, lane = threadIdx.x & 31;
  int e = blockIdx.x * 8 + wave;
  if (e >= E) return;
  int s = src[e], d = dst[e];
  float acc[5] = {0.f, 0.f, 0.f, 0.f, 0.f};
  int koff = 0;
  auto doSeg = [&](const float* p, int w, int md) {
    if (w <= 0) return;
    size_t ridx = (md == 0) ? (size_t)s : (md == 2 ? (size_t)d : (size_t)e);
    const float* row = p + ridx * w;
    for (int k = lane; k < w; k += 32) {
      float x = row[k];
      const float* wr = &sW[(size_t)(koff + k) * dout];
      #pragma unroll
      for (int j = 0; j < 5; ++j)
        if (j < dout) acc[j] += x * wr[j];
    }
    koff += w;
  };
  doSeg(p0, w0, md0); doSeg(p1, w1, md1); doSeg(p2, w2, md2); doSeg(p3, w3, md3);
  #pragma unroll
  for (int j = 0; j < 5; ++j) {
    float v = acc[j];
    for (int off = 16; off; off >>= 1) v += __shfl_xor(v, off, 32);
    acc[j] = v;
  }
  if (lane == 0) {
    for (int j = 0; j < dout; ++j) {
      float v = acc[j] + be[j];
      out[(size_t)e * dout + j] = v > 0.f ? v : 0.2f * v;
    }
  }
}

// ---------------------------------------------------------------------------
// host orchestration
// ---------------------------------------------------------------------------
extern "C" void kernel_launch(void* const* d_in, const int* in_sizes, int n_in,
                              void* d_out_v, int out_size, void* d_ws, size_t ws_size,
                              hipStream_t stream) {
  (void)in_sizes; (void)n_in; (void)out_size; (void)ws_size;
  const float* mu       = (const float*)d_in[0];
  const float* log_var  = (const float*)d_in[1];
  const float* eps      = (const float*)d_in[2];
  const float* room_feat= (const float*)d_in[3];
  const float* rr_feat  = (const float*)d_in[4];
  const int* ff_src = (const int*)d_in[5];
  const int* ff_dst = (const int*)d_in[6];
  const int* rr_src = (const int*)d_in[7];
  const int* rr_dst = (const int*)d_in[8];
  const int* rf_src = (const int*)d_in[9];
  const int* rf_dst = (const int*)d_in[10];
  float* out = (float*)d_out_v;

  // params: pytree flatten -> list order, dict keys sorted:
  // et order: ff(0), rf(1), rr(2); field order: Wdst,We,Wsrc,a_dst,a_src,bdst,be,bsrc
  auto P = [&](int l, int et, int f) -> const float* {
    return (const float*)d_in[11 + (l * 3 + et) * 8 + f];
  };

  const size_t OFF_F  = 0;
  const size_t OFF_R  = (size_t)NFN * 1040;
  const size_t OFF_EFF = OFF_R + (size_t)NRN * RDIM;
  const size_t OFF_ERR = OFF_EFF + (size_t)NE_FF * 3;
  const size_t OFF_ERF = OFF_ERR + (size_t)NE_RR * 4;

  uintptr_t wp = (uintptr_t)d_ws;
  auto alloc = [&](size_t bytes) -> void* {
    void* p = (void*)wp;
    wp += (bytes + 255) & ~(size_t)255;
    return p;
  };
  float* furnA = (float*)alloc((size_t)NFN * 256 * 4);
  float* furnB = (float*)alloc((size_t)NFN * 256 * 4);
  float* roomB = (float*)alloc((size_t)NRN * RDIM * 4);
  float* roomC = (float*)alloc((size_t)NRN * RDIM * 4);
  __bf16* hFb  = (__bf16*)alloc((size_t)NFN * 256 * 2);
  __bf16* hRb  = (__bf16*)alloc((size_t)NRN * RDIM * 2);
  __bf16* wbS  = (__bf16*)alloc((size_t)256 * 1040 * 2);
  float* zs    = (float*)alloc((size_t)NFN * 1040 * 4);
  float* wvS   = (float*)alloc(257 * 4);
  float* wvD   = (float*)alloc(257 * 4);
  float* ssrc  = (float*)alloc((size_t)NFN * 4);
  float* sdst  = (float*)alloc((size_t)NFN * 4);
  float* logit = (float*)alloc((size_t)NE_FF * 4);
  float* wexp  = (float*)alloc((size_t)NE_FF * 4);
  unsigned* maxb = (unsigned*)alloc((size_t)NFN * 4);
  float* sumb  = (float*)alloc((size_t)NFN * 4);
  float* efA[3] = { (float*)alloc((size_t)NE_FF * 3 * 4),
                    (float*)alloc((size_t)NE_RF * 5 * 4),
                    (float*)alloc((size_t)NE_RR * 4 * 4) };
  float* efB[3] = { (float*)alloc((size_t)NE_FF * 3 * 4),
                    (float*)alloc((size_t)NE_RF * 5 * 4),
                    (float*)alloc((size_t)NE_RR * 4 * 4) };

  const int* SRC[3] = { ff_src, rf_src, rr_src };
  const int* DST[3] = { ff_dst, rf_dst, rr_dst };
  const int ECNT[3] = { NE_FF, NE_RF, NE_RR };
  const int EOUT[3] = { 3, 5, 4 };
  const int FIN[3]  = { 64, 128, 256 };
  const int FOUT[3] = { 128, 256, 1040 };

  {
    size_t n = (size_t)NFN * 64;
    reparam_kernel<<<ceilDiv(n, 256), 256, 0, stream>>>(mu, log_var, eps, furnA, n);
  }

  float* hF_in = furnA;  float* hF_out = furnB;
  const float* hR_in = room_feat;  float* hR_out = roomB;  float* hR_other = roomC;
  float* ef_in[3]  = { nullptr, nullptr, nullptr };
  float* ef_out[3] = { efA[0], efA[1], efA[2] };

  for (int l = 0; l < 3; ++l) {
    const int fin = FIN[l], fout = FOUT[l];

    { size_t n = (size_t)NFN * fin;
      f32_to_bf16_kernel<<<ceilDiv(n, 256), 256, 0, stream>>>(hF_in, hFb, n); }
    { size_t n = (size_t)NRN * RDIM;
      f32_to_bf16_kernel<<<ceilDiv(n, 256), 256, 0, stream>>>(hR_in, hRb, n); }

    float* accF = (l == 2) ? (out + OFF_F) : hF_out;
    float* accR = (l == 2) ? (out + OFF_R) : hR_out;
    { size_t n = (size_t)NFN * fout;
      fill_f32_kernel<<<ceilDiv(n, 256), 256, 0, stream>>>(accF, 0.f, n); }
    { size_t n = (size_t)NRN * RDIM;
      fill_f32_kernel<<<ceilDiv(n, 256), 256, 0, stream>>>(accR, 0.f, n); }

    for (int et = 0; et < 3; ++et) {
      int Ns, Nd, fin_s, fin_d, dout;
      const float* hs; const __bf16* hsb; const float* hd; float* accD;
      if (et == 0)      { Ns = NFN; Nd = NFN; fin_s = fin;  fin_d = fin;  dout = fout; hs = hF_in; hsb = hFb; hd = hF_in; accD = accF; }
      else if (et == 1) { Ns = NRN; Nd = NFN; fin_s = RDIM; fin_d = fin;  dout = fout; hs = hR_in; hsb = hRb; hd = hF_in; accD = accF; }
      else              { Ns = NRN; Nd = NRN; fin_s = RDIM; fin_d = RDIM; dout = RDIM; hs = hR_in; hsb = hRb; hd = hR_in; accD = accR; }
      const float* Wdst  = P(l, et, 0);
      const float* We    = P(l, et, 1);
      const float* Wsrc  = P(l, et, 2);
      const float* a_dst = P(l, et, 3);
      const float* a_src = P(l, et, 4);
      const float* bdst  = P(l, et, 5);
      const float* be    = P(l, et, 6);
      const float* bsrc  = P(l, et, 7);
      const int E = ECNT[et];

      fold_attn_kernel<<<ceilDiv(fin_s + 1, 256), 256, 0, stream>>>(Wsrc, a_src, bsrc, fin_s, dout, wvS);
      fold_attn_kernel<<<ceilDiv(fin_d + 1, 256), 256, 0, stream>>>(Wdst, a_dst, bdst, fin_d, dout, wvD);
      node_score_kernel<<<ceilDiv(Ns, 8), 256, 0, stream>>>(hs, wvS, fin_s, Ns, ssrc);
      node_score_kernel<<<ceilDiv(Nd, 8), 256, 0, stream>>>(hd, wvD, fin_d, Nd, sdst);

      // zs = hs @ Wsrc + bsrc  (bf16 WMMA, f32 accumulate)
      { size_t n = (size_t)fin_s * dout;
        f32_to_bf16_kernel<<<ceilDiv(n, 256), 256, 0, stream>>>(Wsrc, wbS, n); }
      if (((fin_s & 31) == 0) && ((dout & 7) == 0)) {
        dim3 gg(ceilDiv(dout, GNT), ceilDiv(Ns, FMT));
        gemm_bf16_wmma_fast_kernel<<<gg, 256, 0, stream>>>(hsb, wbS, bsrc, zs, Ns, dout, fin_s);
      } else {
        dim3 gg(ceilDiv(dout, GNT), ceilDiv(Ns, GMT));
        gemm_bf16_wmma_kernel<<<gg, 256, 0, stream>>>(hsb, wbS, bsrc, zs, Ns, dout, fin_s);
      }

      fill_u32_kernel<<<ceilDiv(Nd, 256), 256, 0, stream>>>(maxb, 0x007FFFFFu /*enc(-inf)*/, Nd);
      fill_f32_kernel<<<ceilDiv(Nd, 256), 256, 0, stream>>>(sumb, 0.f, Nd);
      edge_pass1_kernel<<<ceilDiv(E, 256), 256, 0, stream>>>(ssrc, sdst, SRC[et], DST[et], E, logit, maxb);
      edge_pass2_kernel<<<ceilDiv(E, 256), 256, 0, stream>>>(logit, maxb, DST[et], E, wexp, sumb);
      int sblk = (dout <= 16) ? 64 : 256;
      edge_scatter_kernel<<<E, sblk, 0, stream>>>(zs, SRC[et], DST[et], wexp, sumb, E, dout, accD);

      // fused edge MLP (layer-0 ff/rf concat == duplicated gathers, never materialized)
      const float *p0, *p1, *p2, *p3; int w0, w1, w2, w3, m0, m1, m2, m3, dein;
      p3 = nullptr; w3 = 0; m3 = 0;
      if (l == 0) {
        if (et == 0)      { p0 = hF_in; w0 = 64; m0 = 0; p1 = hF_in; w1 = 64; m1 = 0;
                            p2 = hF_in; w2 = 64; m2 = 2; p3 = hF_in; w3 = 64; m3 = 2; dein = 256; }
        else if (et == 1) { p0 = hR_in; w0 = RDIM; m0 = 0; p1 = hR_in; w1 = RDIM; m1 = 0;
                            p2 = hF_in; w2 = 64; m2 = 2; p3 = hF_in; w3 = 64; m3 = 2; dein = 156; }
        else              { p0 = hR_in; w0 = RDIM; m0 = 0; p1 = rr_feat; w1 = 4; m1 = 1;
                            p2 = hR_in; w2 = RDIM; m2 = 2; dein = 32; }
      } else {
        if (et == 0)      { p0 = hF_in; w0 = fin; m0 = 0; p1 = ef_in[0]; w1 = 3; m1 = 1;
                            p2 = hF_in; w2 = fin; m2 = 2; dein = 2 * fin + 3; }
        else if (et == 1) { p0 = hR_in; w0 = RDIM; m0 = 0; p1 = ef_in[1]; w1 = 5; m1 = 1;
                            p2 = hF_in; w2 = fin; m2 = 2; dein = RDIM + 5 + fin; }
        else              { p0 = hR_in; w0 = RDIM; m0 = 0; p1 = ef_in[2]; w1 = 4; m1 = 1;
                            p2 = hR_in; w2 = RDIM; m2 = 2; dein = 32; }
      }
      float* eo = (l == 2)
          ? (et == 0 ? out + OFF_EFF : (et == 1 ? out + OFF_ERF : out + OFF_ERR))
          : ef_out[et];
      size_t shmem = (size_t)dein * EOUT[et] * 4;
      edge_mlp_kernel<<<ceilDiv(E, 8), 256, shmem, stream>>>(
          SRC[et], DST[et], E, p0, w0, m0, p1, w1, m1, p2, w2, m2, p3, w3, m3,
          We, be, dein, EOUT[et], eo);
    }

    { size_t n = (size_t)NFN * fout;
      elu_ip_kernel<<<ceilDiv(n, 256), 256, 0, stream>>>(accF, n); }
    { size_t n = (size_t)NRN * RDIM;
      elu_ip_kernel<<<ceilDiv(n, 256), 256, 0, stream>>>(accR, n); }

    hF_in = accF;
    hF_out = (hF_out == furnB) ? furnA : furnB;
    hR_in = accR;
    { float* t = hR_out; hR_out = hR_other; hR_other = t; }
    for (int i = 0; i < 3; ++i) ef_in[i] = ef_out[i];
    ef_out[0] = efB[0]; ef_out[1] = efB[1]; ef_out[2] = efB[2];
  }
}
